// NetGraphSage_54683523612722
// MI455X (gfx1250) — matrix operand order repaired and checked by reading, not compiled, not run on hardware
//
#include <hip/hip_runtime.h>
#include <hip/hip_bf16.h>
#include <math.h>

typedef __attribute__((ext_vector_type(16))) _Float16 v16h;
typedef __attribute__((ext_vector_type(8)))  float    v8f;

#define NN   100000
#define EE   3200000
#define FIN  128
#define NB   1000
#define NTILES (NN / 16)          // 6250 exactly

// ---- workspace layout (in floats) ----
#define OFF_AGGR1   ((size_t)0)                       // N*128
#define OFF_DEG     (OFF_AGGR1 + (size_t)NN * 128)    // N
#define OFF_AGGR2   (OFF_DEG   + (size_t)NN)          // N*16
#define OFF_GSUM    (OFF_AGGR2 + (size_t)NN * 16)     // B*16
#define OFF_GCNT    (OFF_GSUM  + (size_t)NB * 16)     // B
#define OFF_H       (OFF_GCNT  + (size_t)NB)          // N*16 (fully overwritten)
#define OFF_END_F32 (OFF_H     + (size_t)NN * 16)
#define ZERO_FLOATS OFF_H                              // zero aggr1..gcnt
// f16 fragment region starts right after f32 region (32B-aligned by construction)
#define W1FRAG_HALF_OFF 0                              // 8 frags * 32 lanes * 16 halves = 4096
#define W2FRAG_HALF_OFF 4096                           // 1 frag  * 32 lanes * 16 halves = 512

// -------------------- prep: build per-lane-swizzled f16 B fragments --------------------
// Fragment layout: wfrag[(kk*32 + lane)*16 + j] = Wpad[kk*32 + (lane>>4)*16 + j][lane&15]
__global__ void k_prep_weights(const float* __restrict__ W1,
                               const float* __restrict__ W2,
                               _Float16* __restrict__ wfrag) {
    int t = blockIdx.x * blockDim.x + threadIdx.x;
    if (t < 4096) {                          // W1 fragments (kk = 0..7)
        int j    = t & 15;
        int lane = (t >> 4) & 31;
        int kk   = t >> 9;
        int n = lane & 15, g = lane >> 4;
        int k = kk * 32 + (g << 4) + j;      // 0..255
        wfrag[W1FRAG_HALF_OFF + t] = (_Float16)((n < 10) ? W1[k * 10 + n] : 0.0f);
    } else if (t < 4096 + 512) {             // W2 fragment (single kk)
        int t2   = t - 4096;
        int j    = t2 & 15;
        int lane = t2 >> 4;
        int n = lane & 15, g = lane >> 4;
        int k = (g << 4) + j;                // 0..31
        wfrag[W2FRAG_HALF_OFF + t2] =
            (_Float16)((k < 20 && n < 10) ? W2[k * 10 + n] : 0.0f);
    }
}

// -------------------- degree & batch counts --------------------
__global__ void k_degree(const int* __restrict__ dst, float* __restrict__ deg) {
    int e = blockIdx.x * blockDim.x + threadIdx.x;
    if (e < EE) atomicAdd(&deg[dst[e]], 1.0f);
}

__global__ void k_batch_count(const int* __restrict__ batch, float* __restrict__ gcnt) {
    int i = blockIdx.x * blockDim.x + threadIdx.x;
    if (i < NN) atomicAdd(&gcnt[batch[i]], 1.0f);
}

// -------------------- scatter layer 1: one wave32 per edge --------------------
__global__ void k_scatter1(const float* __restrict__ x,
                           const int* __restrict__ src,
                           const int* __restrict__ dst,
                           float* __restrict__ aggr1) {
    int wave = (blockIdx.x * blockDim.x + threadIdx.x) >> 5;   // wave-uniform
    if (wave >= EE) return;
    int lane = threadIdx.x & 31;
    int s = src[wave];
    int d = dst[wave];
    float4 v = ((const float4*)(x + (size_t)s * FIN))[lane];    // 128 floats/row
    float* o = aggr1 + (size_t)d * FIN + lane * 4;
    atomicAdd(o + 0, v.x);
    atomicAdd(o + 1, v.y);
    atomicAdd(o + 2, v.z);
    atomicAdd(o + 3, v.w);
}

// -------------------- GEMM1 + ReLU (WMMA f32 <- f16xf16), persistent waves ----------
// h[m][n] = relu( sum_k cat(x, aggr1/deg)[m][k] * W1pad[k][n] ), n in 0..15 (10..15 == 0)
__global__ void k_gemm1_relu(const float* __restrict__ x,
                             const float* __restrict__ aggr1,
                             const float* __restrict__ deg,
                             const _Float16* __restrict__ w1frag,
                             float* __restrict__ h) {
    __shared__ v16h sB[256];                               // 8 frags x 32 lanes, 8 KB
    for (int i = threadIdx.x; i < 256; i += blockDim.x)
        sB[i] = ((const v16h*)w1frag)[i];
    __syncthreads();

    int lane = threadIdx.x & 31;
    int g = lane >> 4;                                     // half-wave select
    int n = lane & 15;                                     // A-row idx AND C/D column
    int nWaves = gridDim.x * (blockDim.x >> 5);
    int wid = blockIdx.x * (blockDim.x >> 5) + (threadIdx.x >> 5);

    // B fragments are tile-invariant: load once into registers (2x ds_load_b128 each)
    v16h bf[8];
#pragma unroll
    for (int kk = 0; kk < 8; ++kk) bf[kk] = sB[kk * 32 + lane];

    for (int tile = wid; tile < NTILES; tile += nWaves) {  // wave-uniform loop
        int m0 = tile * 16;
        size_t rowA = (size_t)(m0 + n) * FIN;
        float rdeg = 1.0f / fmaxf(deg[m0 + n], 1.0f);

        v8f c = {};
#pragma unroll
        for (int kk = 0; kk < 8; ++kk) {                   // K = 256, step 32
            v16h a;
#pragma unroll
            for (int j = 0; j < 16; ++j) {
                // A layout (contiguous, matches B frags): K = 16*g + j + kk*32
                int k = kk * 32 + (g << 4) + j;
                float v = (k < FIN) ? x[rowA + k]
                                    : aggr1[rowA + (k - FIN)] * rdeg;
                a[j] = (_Float16)v;
            }
            c = __builtin_amdgcn_wmma_f32_16x16x32_f16(false, a, false, bf[kk],
                                                       (short)0, c, false, false);
        }
#pragma unroll
        for (int r = 0; r < 8; ++r) {                      // C/D: row = r + 8*g, col = n
            int m = m0 + r + (g << 3);
            h[(size_t)m * 16 + n] = fmaxf(c[r], 0.0f);
        }
    }
}

// -------------------- scatter layer 2: thread per (edge, col), cols 0..9 only -------
__global__ void k_scatter2(const float* __restrict__ h,
                           const int* __restrict__ src,
                           const int* __restrict__ dst,
                           float* __restrict__ aggr2) {
    long long t = (long long)blockIdx.x * blockDim.x + threadIdx.x;
    int e = (int)(t >> 4);
    int cidx = (int)(t & 15);
    if (e < EE && cidx < 10)
        atomicAdd(&aggr2[(size_t)dst[e] * 16 + cidx], h[(size_t)src[e] * 16 + cidx]);
}

// -------------------- GEMM2 + fused global-mean-pool scatter --------------------
__global__ void k_gemm2_pool(const float* __restrict__ h,
                             const float* __restrict__ aggr2,
                             const float* __restrict__ deg,
                             const _Float16* __restrict__ w2frag,
                             const int* __restrict__ batch,
                             float* __restrict__ gsum) {
    __shared__ v16h sB[32];                                // 1 KB
    for (int i = threadIdx.x; i < 32; i += blockDim.x)
        sB[i] = ((const v16h*)w2frag)[i];
    __syncthreads();

    int lane = threadIdx.x & 31;
    int g = lane >> 4;
    int n = lane & 15;
    int nWaves = gridDim.x * (blockDim.x >> 5);
    int wid = blockIdx.x * (blockDim.x >> 5) + (threadIdx.x >> 5);

    v16h b = sB[lane];                                     // tile-invariant fragment

    for (int tile = wid; tile < NTILES; tile += nWaves) {  // wave-uniform loop
        int m0 = tile * 16;
        size_t rowA = (size_t)(m0 + n) * 16;
        float rdeg = 1.0f / fmaxf(deg[m0 + n], 1.0f);

        v16h a;
#pragma unroll
        for (int j = 0; j < 16; ++j) {
            int k = (g << 4) + j;  // 0..31
            float v;
            if (k < 10)      v = h[rowA + k];                    // cat: h part
            else if (k < 20) v = aggr2[rowA + (k - 10)] * rdeg;  // cat: mean aggr
            else             v = 0.0f;                           // pad
            a[j] = (_Float16)v;
        }
        v8f c = {};
        c = __builtin_amdgcn_wmma_f32_16x16x32_f16(false, a, false, b,
                                                   (short)0, c, false, false);
        if (n < 10) {                                      // pool only real columns
#pragma unroll
            for (int r = 0; r < 8; ++r) {
                int m = m0 + r + (g << 3);
                atomicAdd(&gsum[(size_t)batch[m] * 16 + n], c[r]);
            }
        }
    }
}

// -------------------- final: mean, FC, sigmoid --------------------
__global__ void k_final(const float* __restrict__ gsum,
                        const float* __restrict__ gcnt,
                        const float* __restrict__ Wfc,
                        float* __restrict__ out) {
    int b = blockIdx.x * blockDim.x + threadIdx.x;
    if (b >= NB) return;
    float inv = 1.0f / fmaxf(gcnt[b], 1.0f);
    float acc = 0.0f;
#pragma unroll
    for (int d = 0; d < 10; ++d) acc += gsum[(size_t)b * 16 + d] * inv * Wfc[d];
    out[b] = 1.0f / (1.0f + expf(-acc));
}

// ==================== launcher ====================
extern "C" void kernel_launch(void* const* d_in, const int* in_sizes, int n_in,
                              void* d_out, int out_size, void* d_ws, size_t ws_size,
                              hipStream_t stream) {
    const float* x     = (const float*)d_in[0];
    const int*   ei    = (const int*)d_in[1];
    const int*   src   = ei;
    const int*   dst   = ei + EE;
    const int*   batch = (const int*)d_in[2];
    const float* W1    = (const float*)d_in[3];
    const float* W2    = (const float*)d_in[4];
    const float* Wfc   = (const float*)d_in[5];
    float* out = (float*)d_out;

    float* wsf = (float*)d_ws;
    float* aggr1 = wsf + OFF_AGGR1;
    float* deg   = wsf + OFF_DEG;
    float* aggr2 = wsf + OFF_AGGR2;
    float* gsum  = wsf + OFF_GSUM;
    float* gcnt  = wsf + OFF_GCNT;
    float* h     = wsf + OFF_H;
    _Float16* wfrag = (_Float16*)(wsf + OFF_END_F32);

    // zero accumulators (graph-capture safe)
    hipMemsetAsync(d_ws, 0, ZERO_FLOATS * sizeof(float), stream);

    k_prep_weights<<<(4096 + 512 + 255) / 256, 256, 0, stream>>>(W1, W2, wfrag);
    k_degree<<<(EE + 255) / 256, 256, 0, stream>>>(dst, deg);
    k_batch_count<<<(NN + 255) / 256, 256, 0, stream>>>(batch, gcnt);

    // layer 1 aggregation: one wave32 per edge -> 8 edges per 256-thread block
    k_scatter1<<<EE / 8, 256, 0, stream>>>(x, src, dst, aggr1);

    // GEMM1: persistent waves (1024 waves, ~6 tiles each), B frags in registers
    k_gemm1_relu<<<128, 256, 0, stream>>>(x, aggr1, deg,
                                          wfrag + W1FRAG_HALF_OFF, h);

    // layer 2 aggregation: thread per (edge, col), only 10 real columns
    k_scatter2<<<(EE * 16) / 256, 256, 0, stream>>>(h, src, dst, aggr2);

    // GEMM2 + fused mean-pool scatter
    k_gemm2_pool<<<128, 256, 0, stream>>>(h, aggr2, deg,
                                          wfrag + W2FRAG_HALF_OFF, batch, gsum);

    k_final<<<(NB + 255) / 256, 256, 0, stream>>>(gsum, gcnt, Wfc, out);
}
